// GCNEstimator_69836168233269
// MI455X (gfx1250) — compile-verified, matrix operand order checked by reading, not run on hardware
//
#include <hip/hip_runtime.h>
#include <hip/hip_bf16.h>
#include <stdint.h>

typedef __attribute__((ext_vector_type(16))) __bf16 v16bf;
typedef __attribute__((ext_vector_type(8)))  __bf16 v8bf;
typedef __attribute__((ext_vector_type(8)))  float  v8f;

// Fire-and-forget f32 atomic add. gfx1250 has native GLOBAL_ATOMIC_ADD_F32
// (no-return form tracked by STOREcnt; implicit wait at s_endpgm covers us).
// Inline asm guarantees we never fall back to a CAS loop.
__device__ __forceinline__ void atomic_add_f32(float* p, float v) {
#if defined(__gfx1250__)
  asm volatile("global_atomic_add_f32 %0, %1, off" : : "v"(p), "v"(v) : "memory");
#else
  atomicAdd(p, v);
#endif
}

// ---------------- degree / normalization ----------------
__global__ void k_indeg(const int* __restrict__ dst, float* __restrict__ deg, int E) {
  int i = blockIdx.x * blockDim.x + threadIdx.x;
  if (i < E) atomic_add_f32(&deg[dst[i]], 1.0f);
}

__global__ void k_dinv(float* __restrict__ deg, int N) {
  int i = blockIdx.x * blockDim.x + threadIdx.x;
  if (i < N) deg[i] = rsqrtf(deg[i] + 1.0f);  // +1 self loop => always > 0
}

// ---------------- conversions ----------------
__global__ void k_f32_to_bf16(const float* __restrict__ in, __bf16* __restrict__ out, long n) {
  long i = (long)blockIdx.x * blockDim.x + threadIdx.x;
  if (i < n) out[i] = (__bf16)in[i];
}

// W is K x Ncols (row-major); produce Wt as Ncols x K bf16 (each output row = one column of W)
__global__ void k_transpose_to_bf16(const float* __restrict__ W, __bf16* __restrict__ Wt,
                                    int K, int Ncols) {
  int idx = blockIdx.x * blockDim.x + threadIdx.x;
  if (idx < K * Ncols) {
    int n = idx / K, k = idx - n * K;
    Wt[idx] = (__bf16)W[(size_t)k * Ncols + n];
  }
}

__global__ void k_relu_to_bf16(const float* __restrict__ in, __bf16* __restrict__ out, long n) {
  long i = (long)blockIdx.x * blockDim.x + threadIdx.x;
  if (i < n) out[i] = (__bf16)fmaxf(in[i], 0.0f);
}

// agg[n,f] = h[n,f] * dinv[n]^2 + bias[f]   (self-loop message + bias folded in)
__global__ void k_init_agg(const float* __restrict__ h, const float* __restrict__ dinv,
                           const float* __restrict__ bias, float* __restrict__ agg,
                           int N, int F) {
  long i = (long)blockIdx.x * blockDim.x + threadIdx.x;
  if (i < (long)N * F) {
    int n = (int)(i / F);
    int f = (int)(i - (long)n * F);
    float di = dinv[n];
    agg[i] = h[i] * di * di + bias[f];
  }
}

// ---------------- WMMA GEMM: C[M x NC] = A[M x K](bf16) * Bt[NC x K]^T(bf16) ----------------
// One wave computes one 16x16 tile. 16-bit A/B fragment layout (ISA 7.12.2):
// lane<16 -> kb=0, lane>=16 -> kb=8; elements 0..7 at K=kb.., 8..15 at K=16+kb..
template<int K, int NC>
__global__ void k_gemm_wmma_bf16(const __bf16* __restrict__ A, const __bf16* __restrict__ Bt,
                                 float* __restrict__ C, int M) {
  const int lane = threadIdx.x;            // 0..31
  const int l16  = lane & 15;
  const int half = lane >> 4;              // 0 or 1
  const int kb   = half * 8;
  const int mTile = blockIdx.x;
  const int nTile = blockIdx.y * blockDim.y + threadIdx.y;

  int row = mTile * 16 + l16;
  if (row >= M) row = M - 1;               // pad tile: duplicate row, stores guarded
  const int col = nTile * 16 + l16;

  const __bf16* arow = A  + (size_t)row * K;
  const __bf16* brow = Bt + (size_t)col * K;

  v8f acc = {};
  union Frag { v16bf v; v8bf h[2]; };
#pragma unroll
  for (int k0 = 0; k0 < K; k0 += 32) {
    Frag a, b;
    a.h[0] = *(const v8bf*)(arow + k0 + kb);
    a.h[1] = *(const v8bf*)(arow + k0 + 16 + kb);
    b.h[0] = *(const v8bf*)(brow + k0 + kb);
    b.h[1] = *(const v8bf*)(brow + k0 + 16 + kb);
    acc = __builtin_amdgcn_wmma_f32_16x16x32_bf16(
        /*neg_a=*/false, a.v, /*neg_b=*/false, b.v,
        /*c_mod=*/(short)0, acc, /*reuse_a=*/false, /*reuse_b=*/false);
  }

  // C/D layout: lane holds column (lane&15); VGPR r holds row r + 8*half
  const int rbase = mTile * 16 + half * 8;
  const int cidx  = nTile * 16 + l16;
#pragma unroll
  for (int r = 0; r < 8; ++r) {
    int m = rbase + r;
    if (m < M) C[(size_t)m * NC + cidx] = acc[r];
  }
}

// ---------------- edge scatter: agg[dst] += h[src] * dinv[src]*dinv[dst] ----------------
template<int F>
__global__ void k_scatter(const float* __restrict__ h, const int* __restrict__ src,
                          const int* __restrict__ dst, const float* __restrict__ dinv,
                          float* __restrict__ agg, int E) {
  int wid  = (int)(((long)blockIdx.x * blockDim.x + threadIdx.x) >> 5);
  int lane = threadIdx.x & 31;
  if (wid >= E) return;
  int s = src[wid], d = dst[wid];
  float nrm = dinv[s] * dinv[d];
  if constexpr (F == 256) {
    const float4* hp = (const float4*)(h + (size_t)s * F) + lane * 2;
    float4 a = hp[0], b = hp[1];
    float* op = agg + (size_t)d * F + lane * 8;
    atomic_add_f32(op + 0, a.x * nrm); atomic_add_f32(op + 1, a.y * nrm);
    atomic_add_f32(op + 2, a.z * nrm); atomic_add_f32(op + 3, a.w * nrm);
    atomic_add_f32(op + 4, b.x * nrm); atomic_add_f32(op + 5, b.y * nrm);
    atomic_add_f32(op + 6, b.z * nrm); atomic_add_f32(op + 7, b.w * nrm);
  } else {  // F == 64
    const float2* hp = (const float2*)(h + (size_t)s * F) + lane;
    float2 a = *hp;
    float* op = agg + (size_t)d * F + lane * 2;
    atomic_add_f32(op + 0, a.x * nrm);
    atomic_add_f32(op + 1, a.y * nrm);
  }
}

// ---------------- driver ----------------
extern "C" void kernel_launch(void* const* d_in, const int* in_sizes, int n_in,
                              void* d_out, int out_size, void* d_ws, size_t ws_size,
                              hipStream_t stream) {
  const float* x  = (const float*)d_in[0];
  const int*   ei = (const int*)d_in[1];
  const float* W1 = (const float*)d_in[2];
  const float* b1 = (const float*)d_in[3];
  const float* W2 = (const float*)d_in[4];
  const float* b2 = (const float*)d_in[5];

  const int FH = in_sizes[3];            // 256
  const int FO = in_sizes[5];            // 64
  const int FI = in_sizes[2] / FH;       // 512
  const int N  = in_sizes[0] / FI;       // 50000
  const int E  = in_sizes[1] / 2;        // 800000
  const int* src = ei;
  const int* dst = ei + E;

  auto alignup = [](size_t v) { return (v + 255) & ~(size_t)255; };
  uint8_t* ws = (uint8_t*)d_ws;
  size_t o = 0;
  float*  dinv = (float*)(ws + o);   o = alignup(o + (size_t)N * 4);
  __bf16* W1t  = (__bf16*)(ws + o);  o = alignup(o + (size_t)FI * FH * 2);
  __bf16* W2t  = (__bf16*)(ws + o);  o = alignup(o + (size_t)FH * FO * 2);
  size_t xbf_off = o;
  __bf16* xbf  = (__bf16*)(ws + o);  o = alignup(o + (size_t)N * FI * 2);
  float*  h1   = (float*)(ws + o);   o = alignup(o + (size_t)N * FH * 4);
  float*  agg1 = (float*)(ws + o);   o = alignup(o + (size_t)N * FH * 4);
  float*  h2   = (float*)(ws + o);   o = alignup(o + (size_t)N * FO * 4);
  __bf16* h1bf = (__bf16*)(ws + xbf_off);  // reuse x_bf16 region after GEMM1

  // 1) in-degree -> dinv = rsqrt(deg+1)
  hipMemsetAsync(dinv, 0, (size_t)N * 4, stream);
  k_indeg<<<(E + 255) / 256, 256, 0, stream>>>(dst, dinv, E);
  k_dinv<<<(N + 255) / 256, 256, 0, stream>>>(dinv, N);

  // 2) bf16 operands for the GEMMs
  long nx = (long)N * FI;
  k_f32_to_bf16<<<(int)((nx + 255) / 256), 256, 0, stream>>>(x, xbf, nx);
  k_transpose_to_bf16<<<(FI * FH + 255) / 256, 256, 0, stream>>>(W1, W1t, FI, FH);
  k_transpose_to_bf16<<<(FH * FO + 255) / 256, 256, 0, stream>>>(W2, W2t, FH, FO);

  // 3) h1 = x @ W1   (WMMA bf16 -> f32 accum)
  {
    dim3 blk(32, 4);
    dim3 grd((N + 15) / 16, FH / 64);
    k_gemm_wmma_bf16<512, 256><<<grd, blk, 0, stream>>>(xbf, W1t, h1, N);
  }

  // 4) layer-1 propagation (self-loop + bias folded into init), then scatter
  long n1 = (long)N * FH;
  k_init_agg<<<(int)((n1 + 255) / 256), 256, 0, stream>>>(h1, dinv, b1, agg1, N, FH);
  k_scatter<256><<<(E * 32 + 255) / 256, 256, 0, stream>>>(h1, src, dst, dinv, agg1, E);

  // 5) relu -> bf16 hidden
  k_relu_to_bf16<<<(int)((n1 + 255) / 256), 256, 0, stream>>>(agg1, h1bf, n1);

  // 6) h2 = relu(agg1) @ W2
  {
    dim3 blk(32, 4);
    dim3 grd((N + 15) / 16, FO / 64);
    k_gemm_wmma_bf16<256, 64><<<grd, blk, 0, stream>>>(h1bf, W2t, h2, N);
  }

  // 7) layer-2 propagation straight into d_out (bias folded into init)
  float* out = (float*)d_out;
  long n2 = (long)N * FO;
  k_init_agg<<<(int)((n2 + 255) / 256), 256, 0, stream>>>(h2, dinv, b2, out, N, FO);
  k_scatter<64><<<(E * 32 + 255) / 256, 256, 0, stream>>>(h2, src, dst, dinv, out, E);
}
